// AttentionLayer_90305982365943
// MI455X (gfx1250) — matrix operand order repaired and checked by reading, not compiled, not run on hardware
//
#include <hip/hip_runtime.h>

#define B_   2048
#define L_   200
#define E_   128
#define H_   64
#define NEG_ (-1e30f)

typedef __attribute__((ext_vector_type(8)))  float  v8f;
typedef __attribute__((ext_vector_type(8)))  __bf16 v8bf;
typedef __attribute__((ext_vector_type(16))) __bf16 v16bf;

#define FEAT_STRIDE 520   // 512 + 8 bf16 pad -> conflict-free A-frag ds_load_b128
#define H1_STRIDE   72    // 64 + 8 bf16 pad
#define CHUNK       64
#define NCHUNK      4
#define STAGE_BYTES (CHUNK * E_ * 4)                    // 32768 per buffer

#define OFF_FEAT  0
#define OFF_H1    (OFF_FEAT + CHUNK * FEAT_STRIDE * 2)  // 66560
#define OFF_Q     (OFF_H1 + CHUNK * H1_STRIDE * 2)      // 75776
#define OFF_SC    (OFF_Q + E_ * 4)                      // 76288
#define OFF_RED   (OFF_SC + 256 * 4)                    // 77312
#define OFF_STAGE (OFF_RED + 16 * 4)                    // 77376 (16B aligned)
#define SMEM_SZ   (OFF_STAGE + 2 * STAGE_BYTES)         // 142912 bytes

#define W1_FRAG_DW 16384   // 4 nt * 16 kb * 32 lanes * 8 dwords
#define W2_FRAG_DW 2048    // 4 nt * 2 kb  * 32 lanes * 8 dwords

__device__ __forceinline__ unsigned short f2bf(float f) {
  unsigned u = __float_as_uint(f);
  u += 0x7fffu + ((u >> 16) & 1u);        // round-to-nearest-even
  return (unsigned short)(u >> 16);
}
__device__ __forceinline__ unsigned pack2(float a, float b) {
  return (unsigned)f2bf(a) | ((unsigned)f2bf(b) << 16);
}

__device__ __forceinline__ float blk_max(float v, float* red, int tid) {
#pragma unroll
  for (int m = 16; m > 0; m >>= 1) v = fmaxf(v, __shfl_xor(v, m, 32));
  if ((tid & 31) == 0) red[tid >> 5] = v;
  __syncthreads();
  float r = red[0];
#pragma unroll
  for (int w = 1; w < 8; ++w) r = fmaxf(r, red[w]);
  __syncthreads();
  return r;
}
__device__ __forceinline__ float blk_sum(float v, float* red, int tid) {
#pragma unroll
  for (int m = 16; m > 0; m >>= 1) v += __shfl_xor(v, m, 32);
  if ((tid & 31) == 0) red[tid >> 5] = v;
  __syncthreads();
  float r = 0.f;
#pragma unroll
  for (int w = 0; w < 8; ++w) r += red[w];
  __syncthreads();
  return r;
}

// Issue 8 async b128 copies per thread: next chunk's keys -> LDS stage buffer.
// Addresses are clamped (never predicated) so each wave's ASYNCcnt advances by
// exactly 8 -> s_wait_asynccnt 0x8 precisely drains the previous chunk.
__device__ __forceinline__ void stage_issue(const float* keys, unsigned stage_lds,
                                            unsigned cbase, int tid) {
  const unsigned total = (unsigned)B_ * L_ * E_ * 4u;
#pragma unroll
  for (int i = 0; i < 8; ++i) {
    unsigned p    = (unsigned)(i * 256 + tid);
    unsigned goff = cbase + p * 16u;
    if (goff + 16u > total) goff = total - 16u;   // clamp tail (block 2047, chunk 3)
    unsigned ldst = stage_lds + p * 16u;
    asm volatile("global_load_async_to_lds_b128 %0, %1, %2"
                 :: "v"(ldst), "v"(goff), "s"(keys) : "memory");
  }
}

// Pre-swizzle W1 (512x64) and W2 (64x64) fp32 -> bf16 WMMA B-fragment order.
// B-frag (32x16 bf16): lanes 0-15 hold K=0..7,16..23 (2/dword), lanes 16-31 K=8..15,24..31,
// lane%16 = column N; 8 dwords contiguous per lane.
__global__ void __launch_bounds__(256)
prep_wfrag_kernel(const float* __restrict__ W1, const float* __restrict__ W2,
                  unsigned* __restrict__ ws) {
  int i = blockIdx.x * blockDim.x + threadIdx.x;
  if (i < W1_FRAG_DW) {
    int v = i & 7, lane = (i >> 3) & 31, kb = (i >> 8) & 15, nt = i >> 12;
    int K = kb * 32 + ((lane & 16) >> 1) + 2 * (v & 3) + (v & 4) * 4;
    int N = nt * 16 + (lane & 15);
    ws[i] = pack2(W1[K * 64 + N], W1[(K + 1) * 64 + N]);
  }
  if (i < W2_FRAG_DW) {
    int v = i & 7, lane = (i >> 3) & 31, kb = (i >> 8) & 1, nt = (i >> 9) & 3;
    int K = kb * 32 + ((lane & 16) >> 1) + 2 * (v & 3) + (v & 4) * 4;
    int N = nt * 16 + (lane & 15);
    ws[W1_FRAG_DW + i] = pack2(W2[K * 64 + N], W2[(K + 1) * 64 + N]);
  }
}

__global__ void __launch_bounds__(256)
din_attn_kernel(const float* __restrict__ query,
                const float* __restrict__ keys,
                const int*   __restrict__ mask,
                const float* __restrict__ b1,
                const float* __restrict__ b2,
                const float* __restrict__ W3,
                const float* __restrict__ b3,
                const float* __restrict__ nohist,
                const unsigned* __restrict__ wfrag,
                float* __restrict__ out) {
  extern __shared__ char smem[];
  unsigned short* feat = (unsigned short*)(smem + OFF_FEAT);
  unsigned short* h1   = (unsigned short*)(smem + OFF_H1);
  float* qs     = (float*)(smem + OFF_Q);
  float* scores = (float*)(smem + OFF_SC);
  float* red    = (float*)(smem + OFF_RED);

  const int tid  = threadIdx.x;
  const int b    = blockIdx.x;
  const int lane = tid & 31;
  const int wid  = tid >> 5;      // 8 waves
  const int lrow = lane & 15;
  const int lhi  = lane >> 4;     // lane-half: selects K subset of fragments

  // kick off async staging of chunk 0 keys immediately (overlaps with setup)
  const unsigned stage0 = (unsigned)(size_t)(smem + OFF_STAGE);
  stage_issue(keys, stage0, ((unsigned)b * L_) * (E_ * 4u), tid);

  if (tid < E_) qs[tid] = query[(size_t)b * E_ + tid];
  scores[tid] = b3[0];            // init all 256 slots with output bias
  const unsigned* w1f = wfrag;
  const unsigned* w2f = wfrag + W1_FRAG_DW;

  // Both of this wave's tiles share nt -> one register-resident W1/W2 frag set.
  const int nt = wid & 3;
  v16bf breg1[16];
#pragma unroll
  for (int kb = 0; kb < 16; ++kb)
    breg1[kb] = *(const v16bf*)(w1f + (size_t)((nt * 16 + kb) * 32 + lane) * 8);
  v16bf breg2[2];
#pragma unroll
  for (int kb = 0; kb < 2; ++kb)
    breg2[kb] = *(const v16bf*)(w2f + (size_t)((nt * 2 + kb) * 32 + lane) * 8);

  for (int c = 0; c < NCHUNK; ++c) {
    const int lo = c * CHUNK;

    // prefetch next chunk's keys while this chunk computes; drain current chunk
    if (c < NCHUNK - 1) {
      const unsigned sl = (unsigned)(size_t)(smem + OFF_STAGE + ((c + 1) & 1) * STAGE_BYTES);
      stage_issue(keys, sl, ((unsigned)b * L_ + (unsigned)(lo + CHUNK)) * (E_ * 4u), tid);
      asm volatile("s_wait_asynccnt 0x8" ::: "memory");
    } else {
      asm volatile("s_wait_asynccnt 0x0" ::: "memory");
    }
    __syncthreads();

    // ---- build bf16 features [q | k | q*k | q-k] from staged keys ----
    const float* skeys = (const float*)(smem + OFF_STAGE + (c & 1) * STAGE_BYTES);
#pragma unroll 4
    for (int it = 0; it < 16; ++it) {
      int flat = it * 256 + tid;        // 64 rows x 64 element-pairs
      int row  = flat >> 6;
      int ep   = (flat & 63) << 1;
      float q0 = 0.f, q1 = 0.f, k0 = 0.f, k1 = 0.f;
      if (lo + row < L_) {
        k0 = skeys[(row << 7) + ep];
        k1 = skeys[(row << 7) + ep + 1];
        q0 = qs[ep]; q1 = qs[ep + 1];
      }
      unsigned short* fr = feat + row * FEAT_STRIDE;
      *(unsigned*)(fr + ep)          = pack2(q0, q1);
      *(unsigned*)(fr + E_ + ep)     = pack2(k0, k1);
      *(unsigned*)(fr + 2 * E_ + ep) = pack2(q0 * k0, q1 * k1);
      *(unsigned*)(fr + 3 * E_ + ep) = pack2(q0 - k0, q1 - k1);
    }
    __syncthreads();

    // ---- GEMM1: h1 = relu(feat @ W1 + b1)  (64x512x64, bf16 WMMA) ----
    // K-steps grouped by 4: 8 ds_load_b128 issued back-to-back, then 4 WMMAs,
    // so only one dscnt wait is exposed per 4 matrix ops.
#pragma unroll
    for (int tt = 0; tt < 2; ++tt) {
      const int mt = ((wid >> 2) << 1) + tt;
      v8f acc = {};
      const unsigned short* arow = feat + (mt * 16 + lrow) * FEAT_STRIDE + lhi * 8;
#pragma unroll
      for (int kq = 0; kq < 4; ++kq) {
        v8bf lo4[4], hi4[4];
#pragma unroll
        for (int j = 0; j < 4; ++j) {
          lo4[j] = *(const v8bf*)(arow + (kq * 4 + j) * 32);
          hi4[j] = *(const v8bf*)(arow + (kq * 4 + j) * 32 + 16);
        }
#pragma unroll
        for (int j = 0; j < 4; ++j) {
          v16bf a = __builtin_shufflevector(lo4[j], hi4[j],
                      0,1,2,3,4,5,6,7,8,9,10,11,12,13,14,15);
          acc = __builtin_amdgcn_wmma_f32_16x16x32_bf16(
                  false, a, false, breg1[kq * 4 + j], (short)0, acc, false, false);
        }
      }
      const int col = nt * 16 + lrow;
      const float bias = b1[col];
#pragma unroll
      for (int v = 0; v < 8; ++v) {
        float h = fmaxf(acc[v] + bias, 0.f);
        h1[(mt * 16 + v + lhi * 8) * H1_STRIDE + col] = f2bf(h);
      }
    }
    __syncthreads();

    // ---- GEMM2 fused with W3: scores[l] += relu(h1@W2+b2) . W3 ----
#pragma unroll
    for (int tt = 0; tt < 2; ++tt) {
      const int mt = ((wid >> 2) << 1) + tt;
      v8f acc = {};
      const unsigned short* arow = h1 + (mt * 16 + lrow) * H1_STRIDE + lhi * 8;
      v8bf lo2[2], hi2[2];
#pragma unroll
      for (int kb = 0; kb < 2; ++kb) {
        lo2[kb] = *(const v8bf*)(arow + kb * 32);
        hi2[kb] = *(const v8bf*)(arow + kb * 32 + 16);
      }
#pragma unroll
      for (int kb = 0; kb < 2; ++kb) {
        v16bf a = __builtin_shufflevector(lo2[kb], hi2[kb],
                    0,1,2,3,4,5,6,7,8,9,10,11,12,13,14,15);
        acc = __builtin_amdgcn_wmma_f32_16x16x32_bf16(
                false, a, false, breg2[kb], (short)0, acc, false, false);
      }
      const int col  = nt * 16 + lrow;
      const float bias = b2[col];
      const float w3c  = W3[col];
#pragma unroll
      for (int v = 0; v < 8; ++v) {
        float p = fmaxf(acc[v] + bias, 0.f) * w3c;
        p += __shfl_xor(p, 1, 32);
        p += __shfl_xor(p, 2, 32);
        p += __shfl_xor(p, 4, 32);
        p += __shfl_xor(p, 8, 32);       // sum over 16 N-lanes
        if (lrow == 0)
          atomicAdd(&scores[lo + mt * 16 + v + lhi * 8], p);
      }
    }
    __syncthreads();
  }

  // ---- masked softmax over L (matches reference NEG + renormalize math) ----
  float mi = 0.f, si = NEG_;
  if (tid < L_) {
    mi = (float)mask[(size_t)b * L_ + tid];
    if (mi != 0.f) si = scores[tid];
  }
  const float mx  = blk_max(si, red, tid);
  const float ev  = (tid < L_ && mi != 0.f) ? __expf(si - mx) : 0.f;
  const float sum = blk_sum(ev, red, tid);
  const float cnt = blk_sum(mi, red, tid);
  const float at  = ev / (sum + 1e-12f);
  __syncthreads();
  scores[tid] = at;                 // scores now holds attention weights
  __syncthreads();
  const bool all_pad = (cnt == 0.f);

  // ---- user_interest = attn @ keys  (re-read is per-block L2-hot, coalesced) ----
  const int e = tid & (E_ - 1);
  const int half = tid >> 7;        // split L range across two thread-halves
  const float* kp = keys + (size_t)b * L_ * E_ + e;
  float acc = 0.f;
  for (int l = half * 100; l < half * 100 + 100; ++l)
    acc += scores[l] * kp[(size_t)l * E_];
  float* part = (float*)smem;       // reuse feat region
  part[tid] = acc;
  __syncthreads();
  if (tid < E_) {
    float tot = part[tid] + part[tid + 128];
    out[(size_t)b * E_ + tid] = all_pad ? nohist[tid] : tot;
  }
}

extern "C" void kernel_launch(void* const* d_in, const int* in_sizes, int n_in,
                              void* d_out, int out_size, void* d_ws, size_t ws_size,
                              hipStream_t stream) {
  const float* query = (const float*)d_in[0];
  const float* keys  = (const float*)d_in[1];
  const int*   mask  = (const int*)  d_in[2];
  const float* W1    = (const float*)d_in[3];
  const float* b1    = (const float*)d_in[4];
  const float* W2    = (const float*)d_in[5];
  const float* b2    = (const float*)d_in[6];
  const float* W3    = (const float*)d_in[7];
  const float* b3    = (const float*)d_in[8];
  const float* noh   = (const float*)d_in[9];
  unsigned* wfrag    = (unsigned*)d_ws;   // needs (16384+2048)*4 = 73728 B
  float* out         = (float*)d_out;

  prep_wfrag_kernel<<<64, 256, 0, stream>>>(W1, W2, wfrag);
  din_attn_kernel<<<B_, 256, SMEM_SZ, stream>>>(query, keys, mask,
                                                b1, b2, W3, b3, noh, wfrag, out);
}
